// FlaxMistralAttention_33320356282886
// MI455X (gfx1250) — matrix operand order repaired and checked by reading, not compiled, not run on hardware
//
#include <hip/hip_runtime.h>
#include <stdint.h>
#include <stddef.h>

// ---------------------------------------------------------------------------
// Problem constants (match reference)
// ---------------------------------------------------------------------------
#define B_    2
#define S_    2048
#define HID_  4096
#define NH_   32
#define NKV_  8
#define HD_   128
#define WIN_  1024
#define NREP_ (NH_ / NKV_)

typedef __bf16 bf16;
typedef __attribute__((ext_vector_type(16))) __bf16 v16bf;
typedef __attribute__((ext_vector_type(8)))  float  v8f;

// ---------------------------------------------------------------------------
// WMMA fragment load helpers (CDNA5 v_wmma_f32_16x16x32_bf16 layouts)
//
// A-matrix 16x32 bf16: lane L holds row M=L%16.
//   lanes 0-15 : K = {0..7} U {16..23}; lanes 16-31: K = {8..15} U {24..31}
// -> two 16B runs at +(half?16:0) bytes and +32 more bytes.
//
// B-matrix 32x16 bf16: lane L holds column N=L%16.
//   lanes 0-15 : K = {0..7} U {16..23}; lanes 16-31: K = {8..15} U {24..31}
// -> two 16B runs at +(half?16:0) bytes and +32 more bytes.
// ---------------------------------------------------------------------------
__device__ __forceinline__ uint4 ld_g128(const bf16* base, uint32_t byte_off) {
  // uniform base + zero-extended 32-bit offset -> global_load_b128 saddr form
  return *reinterpret_cast<const uint4*>(
      reinterpret_cast<const char*>(base) + byte_off);
}

__device__ __forceinline__ v16bf frag_a_off(const bf16* base, uint32_t off) {
  union { uint4 u[2]; v16bf v; } t;
  t.u[0] = ld_g128(base, off);
  t.u[1] = ld_g128(base, off + 32);
  return t.v;
}

__device__ __forceinline__ v16bf frag_b_off(const bf16* base, uint32_t off) {
  union { uint4 u[2]; v16bf v; } t;
  t.u[0] = ld_g128(base, off);
  t.u[1] = ld_g128(base, off + 32);
  return t.v;
}

// Pointer-based variants (used where the base itself is per-lane, e.g. LDS)
__device__ __forceinline__ v16bf load_a_frag(const bf16* base, int half) {
  const int off = half ? 8 : 0;
  union { uint4 u[2]; v16bf v; } t;
  t.u[0] = *reinterpret_cast<const uint4*>(base + off);
  t.u[1] = *reinterpret_cast<const uint4*>(base + off + 16);
  return t.v;
}

__device__ __forceinline__ v16bf load_b_frag(const bf16* base, int half) {
  const int off = half ? 8 : 0;
  union { uint4 u[2]; v16bf v; } t;
  t.u[0] = *reinterpret_cast<const uint4*>(base + off);
  t.u[1] = *reinterpret_cast<const uint4*>(base + off + 16);
  return t.v;
}

__device__ __forceinline__ v8f wmma_bf16(v16bf a, v16bf b, v8f c) {
  return __builtin_amdgcn_wmma_f32_16x16x32_bf16(false, a, false, b, (short)0, c,
                                                 false, false);
}

// ---------------------------------------------------------------------------
// Elementwise f32 -> bf16 convert
// ---------------------------------------------------------------------------
__global__ void cvt_bf16_kernel(const float* __restrict__ src,
                                bf16* __restrict__ dst, size_t n) {
  size_t i = (size_t)blockIdx.x * blockDim.x + threadIdx.x;
  if (i < n) dst[i] = (bf16)src[i];
}

// Transpose + convert: src is [rows(K), cols(N)] row-major f32,
// dst is [cols(N), rows(K)] row-major bf16 (so WMMA B-frags load contiguous K).
__global__ void transpose_cvt_kernel(const float* __restrict__ src,
                                     bf16* __restrict__ dst,
                                     int rows, int cols) {
  size_t i = (size_t)blockIdx.x * blockDim.x + threadIdx.x;
  size_t total = (size_t)rows * cols;
  if (i >= total) return;
  int k = (int)(i % rows);
  int n = (int)(i / rows);
  dst[(size_t)n * rows + k] = (bf16)src[(size_t)k * cols + n];
}

// ---------------------------------------------------------------------------
// bf16 WMMA GEMM:  C[M,N] f32 = A[M,K] bf16 x Bt[N,K]^T bf16
// 128 threads = 4 waves per block. Block tile 128(M) x 64(N); each wave owns a
// 32(M) x 64(N) strip: 2 A-fragments x 4 B-fragments -> 8 WMMAs per 12 b128
// loads per 32-K step. Addressing uses uniform SGPR bases + 32-bit per-lane
// byte offsets (saddr form), keeping address VGPR pressure minimal so the
// scheduler can overlap next-step loads with WMMAs without spilling.
// M multiple of 128, N multiple of 64, K multiple of 32.
// ---------------------------------------------------------------------------
__global__ void gemm_bf16_wmma_kernel(const bf16* __restrict__ A,
                                      const bf16* __restrict__ Bt,
                                      float* __restrict__ C,
                                      int Mdim, int Ndim, int Kdim) {
  const int lane = threadIdx.x & 31;
  const int wave = threadIdx.x >> 5;
  const int row  = lane & 15;
  const int half = (lane >> 4) & 1;

  const int ntiles = Ndim >> 6;
  const int bm = (blockIdx.x / ntiles) * 128 + wave * 32;
  const int bn = (blockIdx.x % ntiles) * 64;

  v8f acc[2][4] = {};

  const uint32_t krow = (uint32_t)Kdim * 2;  // bytes per row
  uint32_t aoff0 = (uint32_t)(bm + row) * krow + (half ? 16u : 0u);
  uint32_t aoff1 = aoff0 + 16u * krow;
  uint32_t boff0 = (uint32_t)(bn + row) * krow + (half ? 16u : 0u);
  uint32_t boff1 = boff0 + 16u * krow;
  uint32_t boff2 = boff1 + 16u * krow;
  uint32_t boff3 = boff2 + 16u * krow;

#pragma unroll 2
  for (int kb = 0; kb < Kdim; kb += 32) {
    v16bf a0 = frag_a_off(A, aoff0);
    v16bf a1 = frag_a_off(A, aoff1);
    v16bf b0 = frag_b_off(Bt, boff0);
    v16bf b1 = frag_b_off(Bt, boff1);
    v16bf b2 = frag_b_off(Bt, boff2);
    v16bf b3 = frag_b_off(Bt, boff3);

    acc[0][0] = wmma_bf16(a0, b0, acc[0][0]);
    acc[1][0] = wmma_bf16(a1, b0, acc[1][0]);
    acc[0][1] = wmma_bf16(a0, b1, acc[0][1]);
    acc[1][1] = wmma_bf16(a1, b1, acc[1][1]);
    acc[0][2] = wmma_bf16(a0, b2, acc[0][2]);
    acc[1][2] = wmma_bf16(a1, b2, acc[1][2]);
    acc[0][3] = wmma_bf16(a0, b3, acc[0][3]);
    acc[1][3] = wmma_bf16(a1, b3, acc[1][3]);

    aoff0 += 64u; aoff1 += 64u;
    boff0 += 64u; boff1 += 64u; boff2 += 64u; boff3 += 64u;
  }

#pragma unroll
  for (int i = 0; i < 2; ++i) {
#pragma unroll
    for (int j = 0; j < 4; ++j) {
#pragma unroll
      for (int r = 0; r < 8; ++r) {
        int m = bm + i * 16 + r + 8 * half;
        int n = bn + j * 16 + row;
        C[(size_t)m * Ndim + n] = acc[i][j][r];
      }
    }
  }
}

// ---------------------------------------------------------------------------
// RoPE for Q: reads f32 [B*S, NH*HD], writes bf16 [B, NH, S, HD], scale folded.
// ---------------------------------------------------------------------------
__global__ void rope_q_kernel(const float* __restrict__ Qf,
                              const int* __restrict__ pos_ids,
                              bf16* __restrict__ Qb) {
  const float scale = 0.088388347648318447f; // 1/sqrt(128)
  size_t idx = (size_t)blockIdx.x * blockDim.x + threadIdx.x;
  if (idx >= (size_t)B_ * S_ * NH_ * HD_) return;
  int d = (int)(idx & (HD_ - 1));
  size_t t = idx >> 7;
  int h = (int)(t % NH_); t /= NH_;
  int s = (int)(t % S_);
  int b = (int)(t / S_);

  const float* src = Qf + ((size_t)(b * S_ + s) * NH_ + h) * HD_;
  float x = src[d];
  float other = (d < HD_ / 2) ? -src[d + HD_ / 2] : src[d - HD_ / 2];
  float pos = (float)pos_ids[b * S_ + s];
  float inv_freq = __powf(10000.0f, -(2.0f * (float)(d & 63)) / (float)HD_);
  float ang = pos * inv_freq;
  float val = x * cosf(ang) + other * sinf(ang);
  Qb[(((size_t)(b * NH_ + h)) * S_ + s) * HD_ + d] = (bf16)(val * scale);
}

// RoPE for K + layout for V: K -> bf16 [B, NKV, S, HD]; V -> bf16 [B, NKV, HD, S].
__global__ void rope_kv_kernel(const float* __restrict__ Kf,
                               const float* __restrict__ Vf,
                               const int* __restrict__ pos_ids,
                               bf16* __restrict__ Kb,
                               bf16* __restrict__ Vt) {
  size_t idx = (size_t)blockIdx.x * blockDim.x + threadIdx.x;
  if (idx >= (size_t)B_ * S_ * NKV_ * HD_) return;
  int d = (int)(idx & (HD_ - 1));
  size_t t = idx >> 7;
  int h = (int)(t % NKV_); t /= NKV_;
  int s = (int)(t % S_);
  int b = (int)(t / S_);

  size_t src_off = ((size_t)(b * S_ + s) * NKV_ + h) * HD_ + d;
  float x = Kf[src_off];
  size_t src_rot = ((size_t)(b * S_ + s) * NKV_ + h) * HD_ +
                   ((d < HD_ / 2) ? d + HD_ / 2 : d - HD_ / 2);
  float other = (d < HD_ / 2) ? -Kf[src_rot] : Kf[src_rot];
  float pos = (float)pos_ids[b * S_ + s];
  float inv_freq = __powf(10000.0f, -(2.0f * (float)(d & 63)) / (float)HD_);
  float ang = pos * inv_freq;
  float kval = x * cosf(ang) + other * sinf(ang);
  Kb[(((size_t)(b * NKV_ + h)) * S_ + s) * HD_ + d] = (bf16)kval;

  // V transposed so P*V B-fragments load contiguous keys.
  Vt[(((size_t)(b * NKV_ + h)) * HD_ + d) * S_ + s] = (bf16)Vf[src_off];
}

// ---------------------------------------------------------------------------
// Flash attention (sliding window, GQA). One wave per (b, h, 16-query tile).
// 4 waves / block. Emits v_wmma for QK^T and P*V, ds ops for P re-layout.
// ---------------------------------------------------------------------------
__global__ void attn_kernel(const bf16* __restrict__ Qb,
                            const bf16* __restrict__ Kb,
                            const bf16* __restrict__ Vt,
                            const int* __restrict__ amask,
                            bf16* __restrict__ Ob) {
  __shared__ __attribute__((aligned(16))) bf16 pstage[4][16 * 32];

  const int lane = threadIdx.x & 31;
  const int wave = threadIdx.x >> 5;
  const int row  = lane & 15;
  const int half = (lane >> 4) & 1;

  int tile = blockIdx.x * 4 + wave;
  const int QT = S_ / 16;
  int qt = tile % QT; tile /= QT;
  int h = tile % NH_;
  int b = tile / NH_;
  int hk = h / NREP_;
  int qbase = qt * 16;

  // Cache Q A-fragments for the whole tile (HD=128 -> 4 k-steps of 32).
  const bf16* qrow = Qb + (((size_t)(b * NH_ + h)) * S_ + qbase + row) * HD_;
  v16bf qa[4];
#pragma unroll
  for (int t = 0; t < 4; ++t) qa[t] = load_a_frag(qrow + t * 32, half);

  const bf16* Kh = Kb + ((size_t)(b * NKV_ + hk)) * S_ * HD_;
  const bf16* Vh = Vt + ((size_t)(b * NKV_ + hk)) * HD_ * S_;
  const int*  am = amask + (size_t)b * S_;

  float mrun[8], lrun[8];
#pragma unroll
  for (int r = 0; r < 8; ++r) { mrun[r] = -INFINITY; lrun[r] = 0.0f; }
  v8f acc[8] = {};

  int kstart = qbase - WIN_;
  if (kstart < 0) kstart = 0;
  kstart &= ~31;
  const int klast = qbase + 15;

  bf16* pb = pstage[wave];

  for (int kb = kstart; kb <= klast; kb += 32) {
    // ---- scores: S[16q x 32k] via 8 WMMAs -------------------------------
    v8f s0 = {}, s1 = {};
    const bf16* k0 = Kh + (size_t)(kb + row) * HD_;
    const bf16* k1 = Kh + (size_t)(kb + 16 + row) * HD_;
#pragma unroll
    for (int t = 0; t < 4; ++t) {
      v16bf b0 = load_b_frag(k0 + t * 32, half);
      s0 = wmma_bf16(qa[t], b0, s0);
      v16bf b1 = load_b_frag(k1 + t * 32, half);
      s1 = wmma_bf16(qa[t], b1, s1);
    }

    const int kj0 = kb + row;
    const int kj1 = kj0 + 16;
    const bool am0 = am[kj0] > 0;
    const bool am1 = am[kj1] > 0;

    // ---- online softmax (row = r + 8*half, cols striped over 16 lanes) --
    float p0[8], p1[8], alpha[8];
#pragma unroll
    for (int r = 0; r < 8; ++r) {
      int qi = qbase + r + 8 * half;
      float v0 = ((kj0 <= qi) && (kj0 >= qi - WIN_) && am0) ? s0[r] : -1e30f;
      float v1 = ((kj1 <= qi) && (kj1 >= qi - WIN_) && am1) ? s1[r] : -1e30f;
      float rm = fmaxf(v0, v1);
#pragma unroll
      for (int m = 1; m < 16; m <<= 1) rm = fmaxf(rm, __shfl_xor(rm, m, 32));
      float mn = fmaxf(mrun[r], rm);
      float al = __expf(mrun[r] - mn);
      float e0 = __expf(v0 - mn);
      float e1 = __expf(v1 - mn);
      float rs = e0 + e1;
#pragma unroll
      for (int m = 1; m < 16; m <<= 1) rs += __shfl_xor(rs, m, 32);
      lrun[r] = lrun[r] * al + rs;
      mrun[r] = mn;
      p0[r] = e0; p1[r] = e1; alpha[r] = al;
    }

#pragma unroll
    for (int o = 0; o < 8; ++o)
#pragma unroll
      for (int r = 0; r < 8; ++r) acc[o][r] *= alpha[r];

    // ---- restage P (D-layout -> A-fragment layout) through LDS ----------
#pragma unroll
    for (int r = 0; r < 8; ++r) {
      pb[(r + 8 * half) * 32 + row]      = (bf16)p0[r];
      pb[(r + 8 * half) * 32 + 16 + row] = (bf16)p1[r];
    }
    // wave-synchronous LDS exchange: order stores before loads
    asm volatile("s_wait_dscnt 0" ::: "memory");
    v16bf pa = load_a_frag(pb + row * 32, half);
    asm volatile("" ::: "memory");

    // ---- P(16x32) x V(32x128): 8 WMMAs into 8 accumulators --------------
#pragma unroll
    for (int o = 0; o < 8; ++o) {
      const bf16* vrow = Vh + (size_t)(o * 16 + row) * S_ + kb;
      v16bf bv = load_b_frag(vrow, half);
      acc[o] = wmma_bf16(pa, bv, acc[o]);
    }
  }

  // ---- epilogue: O /= l, write bf16 [B, S, NH*HD] -----------------------
#pragma unroll
  for (int o = 0; o < 8; ++o) {
#pragma unroll
    for (int r = 0; r < 8; ++r) {
      int qi = qbase + r + 8 * half;
      float val = acc[o][r] / lrun[r];
      Ob[((size_t)(b * S_ + qi)) * HID_ + h * HD_ + o * 16 + row] = (bf16)val;
    }
  }
}

// ---------------------------------------------------------------------------
// Host-side launch
// ---------------------------------------------------------------------------
extern "C" void kernel_launch(void* const* d_in, const int* in_sizes, int n_in,
                              void* d_out, int out_size, void* d_ws, size_t ws_size,
                              hipStream_t stream) {
  (void)in_sizes; (void)n_in; (void)out_size; (void)ws_size;

  const float* hs    = (const float*)d_in[0];
  const int*   amask = (const int*)d_in[1];
  const int*   posid = (const int*)d_in[2];
  const float* qw    = (const float*)d_in[3];
  const float* kw    = (const float*)d_in[4];
  const float* vw    = (const float*)d_in[5];
  const float* ow    = (const float*)d_in[6];
  float* out = (float*)d_out;

  const size_t M = (size_t)B_ * S_;  // 4096 rows

  char* ws = (char*)d_ws;
  size_t off = 0;
  auto take = [&](size_t bytes) -> char* {
    char* p = ws + off;
    off += (bytes + 255) & ~(size_t)255;
    return p;
  };

  bf16* hsb = (bf16*)take(M * HID_ * 2);                       // 32 MB
  bf16* qwt = (bf16*)take((size_t)HID_ * (NH_ * HD_) * 2);     // 32 MB  [N,K]
  bf16* kwt = (bf16*)take((size_t)HID_ * (NKV_ * HD_) * 2);    //  8 MB
  bf16* vwt = (bf16*)take((size_t)HID_ * (NKV_ * HD_) * 2);    //  8 MB
  bf16* owt = (bf16*)take((size_t)(NH_ * HD_) * HID_ * 2);     // 32 MB
  float* qf = (float*)take(M * (NH_ * HD_) * 4);               // 64 MB (reused below)
  float* kf = (float*)take(M * (NKV_ * HD_) * 4);              // 16 MB
  float* vf = (float*)take(M * (NKV_ * HD_) * 4);              // 16 MB
  bf16* qbf = (bf16*)take(M * (NH_ * HD_) * 2);                // 32 MB
  bf16* kbf = (bf16*)take(M * (NKV_ * HD_) * 2);               //  8 MB
  bf16* vtb = (bf16*)take(M * (NKV_ * HD_) * 2);               //  8 MB
  bf16* attnb = (bf16*)qf;  // qf is dead after rope_q; alias for attn output

  const int T = 256;

  // 1) convert / transpose weights + activations to bf16
  cvt_bf16_kernel<<<(unsigned)((M * HID_ + T - 1) / T), T, 0, stream>>>(hs, hsb, M * HID_);
  transpose_cvt_kernel<<<(unsigned)(((size_t)HID_ * NH_ * HD_ + T - 1) / T), T, 0, stream>>>(
      qw, qwt, HID_, NH_ * HD_);
  transpose_cvt_kernel<<<(unsigned)(((size_t)HID_ * NKV_ * HD_ + T - 1) / T), T, 0, stream>>>(
      kw, kwt, HID_, NKV_ * HD_);
  transpose_cvt_kernel<<<(unsigned)(((size_t)HID_ * NKV_ * HD_ + T - 1) / T), T, 0, stream>>>(
      vw, vwt, HID_, NKV_ * HD_);
  transpose_cvt_kernel<<<(unsigned)(((size_t)(NH_ * HD_) * HID_ + T - 1) / T), T, 0, stream>>>(
      ow, owt, NH_ * HD_, HID_);

  // 2) QKV projections (bf16 WMMA, f32 out). Block tile 128x64, 4 waves.
  gemm_bf16_wmma_kernel<<<(unsigned)((M / 128) * ((NH_ * HD_) / 64)), 128, 0, stream>>>(
      hsb, qwt, qf, (int)M, NH_ * HD_, HID_);
  gemm_bf16_wmma_kernel<<<(unsigned)((M / 128) * ((NKV_ * HD_) / 64)), 128, 0, stream>>>(
      hsb, kwt, kf, (int)M, NKV_ * HD_, HID_);
  gemm_bf16_wmma_kernel<<<(unsigned)((M / 128) * ((NKV_ * HD_) / 64)), 128, 0, stream>>>(
      hsb, vwt, vf, (int)M, NKV_ * HD_, HID_);

  // 3) RoPE + attention-friendly layouts
  rope_q_kernel<<<(unsigned)((M * NH_ * HD_ + T - 1) / T), T, 0, stream>>>(qf, posid, qbf);
  rope_kv_kernel<<<(unsigned)((M * NKV_ * HD_ + T - 1) / T), T, 0, stream>>>(
      kf, vf, posid, kbf, vtb);

  // 4) sliding-window GQA flash attention (one wave per 16-query tile)
  attn_kernel<<<(unsigned)((B_ * NH_ * (S_ / 16)) / 4), 128, 0, stream>>>(
      qbf, kbf, vtb, amask, attnb);

  // 5) output projection -> f32 d_out
  gemm_bf16_wmma_kernel<<<(unsigned)((M / 128) * (HID_ / 64)), 128, 0, stream>>>(
      attnb, owt, out, (int)M, HID_, HID_);
}
